// MultiHeadAttention_26628797235727
// MI455X (gfx1250) — matrix operand order repaired and checked by reading, not compiled, not run on hardware
//
#include <hip/hip_runtime.h>
#include <hip/hip_bf16.h>
#include <stdint.h>

typedef __attribute__((ext_vector_type(16))) _Float16 v16h;
typedef __attribute__((ext_vector_type(8)))  _Float16 v8h;
typedef __attribute__((ext_vector_type(8)))  float    v8f;

#define D_MODEL   1024
#define NUM_HEADS 16
#define HEAD_DIM  64
#define BATCH     4
#define SEQ       2048
#define MROWS     (BATCH * SEQ)   // 8192

// ---------------------------------------------------------------------------
// WMMA helpers (gfx1250: V_WMMA_F32_16X16X32_F16, wave32)
// ---------------------------------------------------------------------------
static __device__ __forceinline__ v8f wmma_f16(v16h a, v16h b, v8f c) {
  return __builtin_amdgcn_wmma_f32_16x16x32_f16(false, a, false, b, (short)0, c,
                                                false, false);
}

// A fragment, 16x32 f16, row-major source with row stride `stride` (halfs).
// Layout used here (contiguous K): lane = m + 16*hi; halfs j -> K = k0 + 16*hi + j
static __device__ __forceinline__ v16h load_a16(const _Float16* base, int stride,
                                                int m, int hi, int k0) {
  const _Float16* p = base + (size_t)m * stride + k0 + 16 * hi;
  v16h a;
  ((v8h*)&a)[0] = *(const v8h*)(p);
  ((v8h*)&a)[1] = *(const v8h*)(p + 8);
  return a;
}

// B fragment, 32x16 f16, column n contiguous in K (source = Bt[n][k], row
// stride `stride`). Layout used here (contiguous K): lane = n + 16*kh; halfs j -> K = k0 + 16*kh + j
static __device__ __forceinline__ v16h load_b16(const _Float16* base, int stride,
                                                int n, int kh, int k0) {
  const _Float16* p = base + (size_t)n * stride + k0 + 16 * kh;
  v16h b;
  ((v8h*)&b)[0] = *(const v8h*)(p);
  ((v8h*)&b)[1] = *(const v8h*)(p + 8);
  return b;
}

// Async copy: 32 B (two b128) global -> LDS, ASYNCcnt-tracked.
// INST_OFFSET applies to both global and LDS addresses (ISA §15.18 async).
static __device__ __forceinline__ void async_copy32(unsigned lds_byte_off,
                                                    const _Float16* gaddr) {
  asm volatile("global_load_async_to_lds_b128 %0, %1, off"
               :: "v"(lds_byte_off), "v"(gaddr) : "memory");
  asm volatile("global_load_async_to_lds_b128 %0, %1, off offset:16"
               :: "v"(lds_byte_off), "v"(gaddr) : "memory");
}

// ---------------------------------------------------------------------------
// Precision conversion kernels
// ---------------------------------------------------------------------------
__global__ void cast_f32_to_f16(const float* __restrict__ in,
                                _Float16* __restrict__ out, int n) {
  int i = blockIdx.x * blockDim.x + threadIdx.x;
  if (i < n) out[i] = (_Float16)in[i];
}

// W[k][n] f32  ->  Wt[n][k] f16   (K = N = 1024), LDS-tiled transpose
__global__ void transpose_cast_w(const float* __restrict__ W,
                                 _Float16* __restrict__ Wt) {
  __shared__ float tile[32][33];
  int k0 = blockIdx.x * 32, n0 = blockIdx.y * 32;
  int tx = threadIdx.x & 31, ty = threadIdx.x >> 5;
  for (int r = ty; r < 32; r += 8)
    tile[r][tx] = W[(size_t)(k0 + r) * D_MODEL + n0 + tx];
  __syncthreads();
  for (int r = ty; r < 32; r += 8)
    Wt[(size_t)(n0 + r) * D_MODEL + k0 + tx] = (_Float16)tile[tx][r];
}

// ---------------------------------------------------------------------------
// Block-tiled WMMA GEMM: C[M,N] = (A[M,K] * Bt[N,K]^T + bias) * scale
// Block = 256 threads = 8 waves, computes a 128x128 C tile.
// Wave grid 2(m) x 4(n); each wave: 64x32 tile = 4x2 fragments.
// A/B K-step tiles (128x32 f16 each) staged in LDS via async-to-LDS copies,
// double buffered, drained with s_wait_asynccnt.
//   mode 0: out16 head-split  [b,h,s,dh]   (Q with scale=1/8, K with scale=1)
//   mode 1: out16 head-transposed [b,h,dh,s]  (V)
//   mode 2: out32 row-major [M,N] f32      (final output projection)
// ---------------------------------------------------------------------------
__global__ void gemm_wmma(const _Float16* __restrict__ A,
                          const _Float16* __restrict__ Bt,
                          const float* __restrict__ bias, float scale, int mode,
                          _Float16* __restrict__ out16,
                          float* __restrict__ out32) {
  const int K = D_MODEL, N = D_MODEL;
  __shared__ __align__(16) _Float16 Atile[2][128 * 32];  // 8 KiB x2
  __shared__ __align__(16) _Float16 Btile[2][128 * 32];  // 8 KiB x2

  int tid  = threadIdx.x;
  int lane = tid & 31, wave = tid >> 5;
  int wm = wave >> 2, wn = wave & 3;  // 2 x 4 wave grid
  int ml = lane & 15, hi = lane >> 4;

  const int tiles_n = N / 128;
  int m0 = (blockIdx.x / tiles_n) * 128;
  int n0 = (blockIdx.x % tiles_n) * 128;

  // Per-thread staging assignment: 128 rows x 32 halfs per tile,
  // thread t copies 16 contiguous halfs of row t/2 at col (t&1)*16.
  int srow = tid >> 1;
  int scol = (tid & 1) * 16;
  const _Float16* gA = A + ((size_t)(m0 + srow)) * K + scol;
  const _Float16* gB = Bt + ((size_t)(n0 + srow)) * K + scol;
  unsigned lA0 = (unsigned)(uintptr_t)(&Atile[0][srow * 32 + scol]);
  unsigned lA1 = (unsigned)(uintptr_t)(&Atile[1][srow * 32 + scol]);
  unsigned lB0 = (unsigned)(uintptr_t)(&Btile[0][srow * 32 + scol]);
  unsigned lB1 = (unsigned)(uintptr_t)(&Btile[1][srow * 32 + scol]);

  v8f acc[4][2];
#pragma unroll
  for (int mf = 0; mf < 4; ++mf)
#pragma unroll
    for (int nf = 0; nf < 2; ++nf) { v8f z = {}; acc[mf][nf] = z; }

  // Prologue: stage k-step 0 into buffer 0.
  async_copy32(lA0, gA);
  async_copy32(lB0, gB);

  int buf = 0;
  for (int k0 = 0; k0 < K; k0 += 32) {
    // Prefetch next k-step into the other buffer.
    if (k0 + 32 < K) {
      if (buf == 0) {
        async_copy32(lA1, gA + k0 + 32);
        async_copy32(lB1, gB + k0 + 32);
      } else {
        async_copy32(lA0, gA + k0 + 32);
        async_copy32(lB0, gB + k0 + 32);
      }
      asm volatile("s_wait_asynccnt 0x4" ::: "memory");  // current buf done
    } else {
      asm volatile("s_wait_asynccnt 0x0" ::: "memory");
    }
    __syncthreads();

    const _Float16* At = &Atile[buf][0];
    const _Float16* Bl = &Btile[buf][0];
    v16h af[4], bf[2];
#pragma unroll
    for (int mf = 0; mf < 4; ++mf)
      af[mf] = load_a16(At + (wm * 64 + mf * 16) * 32, 32, ml, hi, 0);
#pragma unroll
    for (int nf = 0; nf < 2; ++nf)
      bf[nf] = load_b16(Bl + (wn * 32 + nf * 16) * 32, 32, ml, hi, 0);
#pragma unroll
    for (int mf = 0; mf < 4; ++mf)
#pragma unroll
      for (int nf = 0; nf < 2; ++nf)
        acc[mf][nf] = wmma_f16(af[mf], bf[nf], acc[mf][nf]);

    __syncthreads();  // everyone done reading `buf` before it is restaged
    buf ^= 1;
  }

  // Epilogue
#pragma unroll
  for (int nf = 0; nf < 2; ++nf) {
    int col = n0 + wn * 32 + nf * 16 + ml;
    float bv = bias[col];
    int h = col >> 6, dh = col & 63;
#pragma unroll
    for (int mf = 0; mf < 4; ++mf) {
#pragma unroll
      for (int r = 0; r < 8; ++r) {
        int row = m0 + wm * 64 + mf * 16 + r + 8 * hi;
        float val = (acc[mf][nf][r] + bv) * scale;
        if (mode == 2) {
          out32[(size_t)row * N + col] = val;
        } else {
          int bb = row / SEQ, s = row % SEQ;
          if (mode == 0)
            out16[(((size_t)bb * NUM_HEADS + h) * SEQ + s) * HEAD_DIM + dh] =
                (_Float16)val;
          else
            out16[(((size_t)bb * NUM_HEADS + h) * HEAD_DIM + dh) * SEQ + s] =
                (_Float16)val;
        }
      }
    }
  }
}

// ---------------------------------------------------------------------------
// Flash attention: one wave (block of 32) per (b, h, 16-query tile).
// Q pre-scaled by 1/sqrt(Dh). Streams keys in chunks of 32 with online
// softmax; P tile bounced through LDS to re-swizzle C-layout -> A-layout.
// ---------------------------------------------------------------------------
__global__ void attn_flash(const _Float16* __restrict__ Qh,
                           const _Float16* __restrict__ Kh,
                           const _Float16* __restrict__ Vt,
                           _Float16* __restrict__ Ao) {
  __shared__ __align__(16) _Float16 Plds[16 * 32];
  int lane = threadIdx.x;
  int m = lane & 15, hi = lane >> 4;
  const int qtiles = SEQ / 16;
  int bh = blockIdx.x / qtiles;
  int q0 = (blockIdx.x % qtiles) * 16;
  int b = bh / NUM_HEADS, h = bh % NUM_HEADS;

  const _Float16* Qbase = Qh + ((size_t)bh * SEQ + q0) * HEAD_DIM;
  const _Float16* Kbase = Kh + (size_t)bh * SEQ * HEAD_DIM;
  const _Float16* Vbase = Vt + (size_t)bh * HEAD_DIM * SEQ;

  v16h qA[2];
#pragma unroll
  for (int s = 0; s < 2; ++s) qA[s] = load_a16(Qbase, HEAD_DIM, m, hi, 32 * s);

  float runmax[8], runsum[8], ef[8];
  v8f O[4];
#pragma unroll
  for (int r = 0; r < 8; ++r) { runmax[r] = -3.0e38f; runsum[r] = 0.f; }
#pragma unroll
  for (int t = 0; t < 4; ++t) { v8f z = {}; O[t] = z; }

  for (int kc = 0; kc < SEQ; kc += 32) {
    v8f sc0 = {}, sc1 = {};
#pragma unroll
    for (int s = 0; s < 2; ++s) {
      v16h b0 = load_b16(Kbase + (size_t)kc * HEAD_DIM, HEAD_DIM, m, hi, 32 * s);
      v16h b1 =
          load_b16(Kbase + (size_t)(kc + 16) * HEAD_DIM, HEAD_DIM, m, hi, 32 * s);
      sc0 = wmma_f16(qA[s], b0, sc0);
      sc1 = wmma_f16(qA[s], b1, sc1);
    }

#pragma unroll
    for (int r = 0; r < 8; ++r) {
      float v = fmaxf(sc0[r], sc1[r]);
#pragma unroll
      for (int off = 1; off < 16; off <<= 1) v = fmaxf(v, __shfl_xor(v, off, 32));
      float nm = fmaxf(runmax[r], v);
      ef[r] = __expf(runmax[r] - nm);
      float p0 = __expf(sc0[r] - nm);
      float p1 = __expf(sc1[r] - nm);
      float rs = p0 + p1;
#pragma unroll
      for (int off = 1; off < 16; off <<= 1) rs += __shfl_xor(rs, off, 32);
      runsum[r] = runsum[r] * ef[r] + rs;
      runmax[r] = nm;
      Plds[(r + 8 * hi) * 32 + m] = (_Float16)p0;
      Plds[(r + 8 * hi) * 32 + 16 + m] = (_Float16)p1;
    }
    __syncthreads();

    v16h pA = load_a16(&Plds[0], 32, m, hi, 0);
#pragma unroll
    for (int t = 0; t < 4; ++t) {
#pragma unroll
      for (int r = 0; r < 8; ++r) O[t][r] *= ef[r];
      v16h vB;
      const _Float16* p = Vbase + (size_t)(16 * t + m) * SEQ + kc + 16 * hi;
      ((v8h*)&vB)[0] = *(const v8h*)(p);
      ((v8h*)&vB)[1] = *(const v8h*)(p + 8);
      O[t] = wmma_f16(pA, vB, O[t]);
    }
    __syncthreads();
  }

#pragma unroll
  for (int r = 0; r < 8; ++r) runsum[r] = 1.f / runsum[r];
#pragma unroll
  for (int t = 0; t < 4; ++t) {
#pragma unroll
    for (int r = 0; r < 8; ++r) {
      int row = b * SEQ + q0 + r + 8 * hi;
      int col = h * HEAD_DIM + 16 * t + m;
      Ao[(size_t)row * D_MODEL + col] = (_Float16)(O[t][r] * runsum[r]);
    }
  }
}

// ---------------------------------------------------------------------------
// Host launcher
// ---------------------------------------------------------------------------
extern "C" void kernel_launch(void* const* d_in, const int* in_sizes, int n_in,
                              void* d_out, int out_size, void* d_ws,
                              size_t ws_size, hipStream_t stream) {
  const float* q  = (const float*)d_in[0];
  const float* k  = (const float*)d_in[1];
  const float* v  = (const float*)d_in[2];
  const float* Wq = (const float*)d_in[3];
  const float* bq = (const float*)d_in[4];
  const float* Wk = (const float*)d_in[5];
  const float* bk = (const float*)d_in[6];
  const float* Wv = (const float*)d_in[7];
  const float* bv = (const float*)d_in[8];
  const float* Wo = (const float*)d_in[9];
  const float* bo = (const float*)d_in[10];
  float* out = (float*)d_out;

  const size_t ACT = (size_t)MROWS * D_MODEL * sizeof(_Float16);
  const size_t WT  = (size_t)D_MODEL * D_MODEL * sizeof(_Float16);
  size_t need = 3 * ACT + 4 * WT + 3 * ACT + ACT;
  if (ws_size < need) return;

  char* ws = (char*)d_ws;
  size_t off = 0;
  auto alloc = [&](size_t bytes) {
    void* p = ws + off;
    off = (off + bytes + 255) & ~(size_t)255;
    return p;
  };
  _Float16* Xq  = (_Float16*)alloc(ACT);
  _Float16* Xk  = (_Float16*)alloc(ACT);
  _Float16* Xv  = (_Float16*)alloc(ACT);
  _Float16* WqT = (_Float16*)alloc(WT);
  _Float16* WkT = (_Float16*)alloc(WT);
  _Float16* WvT = (_Float16*)alloc(WT);
  _Float16* WoT = (_Float16*)alloc(WT);
  _Float16* Qh  = (_Float16*)alloc(ACT);  // [b,h,s,dh] (pre-scaled by 1/8)
  _Float16* Kh  = (_Float16*)alloc(ACT);  // [b,h,s,dh]
  _Float16* Vt  = (_Float16*)alloc(ACT);  // [b,h,dh,s]
  _Float16* Ao  = (_Float16*)alloc(ACT);  // [b*s, d_model]

  const int nElem = MROWS * D_MODEL;
  const int castBlocks = (nElem + 255) / 256;
  cast_f32_to_f16<<<castBlocks, 256, 0, stream>>>(q, Xq, nElem);
  cast_f32_to_f16<<<castBlocks, 256, 0, stream>>>(k, Xk, nElem);
  cast_f32_to_f16<<<castBlocks, 256, 0, stream>>>(v, Xv, nElem);

  dim3 tg(D_MODEL / 32, D_MODEL / 32);
  transpose_cast_w<<<tg, 256, 0, stream>>>(Wq, WqT);
  transpose_cast_w<<<tg, 256, 0, stream>>>(Wk, WkT);
  transpose_cast_w<<<tg, 256, 0, stream>>>(Wv, WvT);
  transpose_cast_w<<<tg, 256, 0, stream>>>(Wo, WoT);

  const int gemmBlocks = (MROWS / 128) * (D_MODEL / 128);  // 512
  const float invSqrtDh = 0.125f;                          // 1/sqrt(64)
  gemm_wmma<<<gemmBlocks, 256, 0, stream>>>(Xq, WqT, bq, invSqrtDh, 0, Qh, nullptr);
  gemm_wmma<<<gemmBlocks, 256, 0, stream>>>(Xk, WkT, bk, 1.0f, 0, Kh, nullptr);
  gemm_wmma<<<gemmBlocks, 256, 0, stream>>>(Xv, WvT, bv, 1.0f, 1, Vt, nullptr);

  const int attnBlocks = BATCH * NUM_HEADS * (SEQ / 16);  // 8192
  attn_flash<<<attnBlocks, 32, 0, stream>>>(Qh, Kh, Vt, Ao);

  gemm_wmma<<<gemmBlocks, 256, 0, stream>>>(Ao, WoT, bo, 1.0f, 2, nullptr, out);
}
